// LabelAttention_4423816314978
// MI455X (gfx1250) — matrix operand order repaired and checked by reading, not compiled, not run on hardware
//
#include <hip/hip_runtime.h>
#include <hip/hip_bf16.h>

// Problem constants (match reference)
#define BB   16
#define SS   512
#define LL   4096
#define HH   512
#define DIN  1024
#define DLBL 768

typedef __attribute__((ext_vector_type(16))) __bf16       v16bf;
typedef __attribute__((ext_vector_type(4)))  __bf16       v4bf;
typedef __attribute__((ext_vector_type(8)))  float        v8f;
typedef __attribute__((ext_vector_type(8)))  unsigned int v8u;
typedef __attribute__((ext_vector_type(4)))  unsigned int v4u;

// Combine two 16-byte halves into a 16x32 bf16 A/B fragment register block.
__device__ __forceinline__ v16bf frag_combine(v4u lo, v4u hi) {
  v8u u;
  u[0] = lo[0]; u[1] = lo[1]; u[2] = lo[2]; u[3] = lo[3];
  u[4] = hi[0]; u[5] = hi[1]; u[6] = hi[2]; u[7] = hi[3];
  return __builtin_bit_cast(v16bf, u);
}

// A-operand (16-bit A 16x32 layout): lane row = ln, halves at K offsets
//   VGPR0..3 : K = 8*half .. 8*half+7      (16 contiguous bytes)
//   VGPR4..7 : K = 16+8*half .. 16+8*half+7 (16 contiguous bytes)
__device__ __forceinline__ v16bf load_afrag(const __bf16* rowk, int half) {
  v4u lo = *(const v4u*)(rowk + 8 * half);
  v4u hi = *(const v4u*)(rowk + 16 + 8 * half);
  return frag_combine(lo, hi);
}

// B-operand (16-bit B 32x16 layout): lane col = ln, same K order as A:
// K = 8*half..8*half+7 and 16+8*half..16+8*half+7 (two b128s).
__device__ __forceinline__ v16bf load_bfrag(const __bf16* colk, int half) {
  v4u lo = *(const v4u*)(colk + 8 * half);
  v4u hi = *(const v4u*)(colk + 16 + 8 * half);
  return frag_combine(lo, hi);
}

// ---------------------------------------------------------------------------
// f32 -> bf16 elementwise convert, 4 elements/thread/iter (b128 in, b64 out)
// ---------------------------------------------------------------------------
__global__ void cvt_f32_bf16_x4(const float* __restrict__ src,
                                __bf16* __restrict__ dst, int n4) {
  int i = blockIdx.x * blockDim.x + threadIdx.x;
  int stride = gridDim.x * blockDim.x;
  for (; i < n4; i += stride) {
    float4 f = ((const float4*)src)[i];
    v4bf r = { (__bf16)f.x, (__bf16)f.y, (__bf16)f.z, (__bf16)f.w };
    *(v4bf*)(dst + 4 * (size_t)i) = r;
  }
}

// ---------------------------------------------------------------------------
// Tiled transpose + convert: inputs f32 [B][S][DIN] -> Vt bf16 [B][DIN][S]
// 32x32 tiles, coalesced reads and writes through LDS.
// ---------------------------------------------------------------------------
__global__ __launch_bounds__(256) void transpose_to_bf16(
    const float* __restrict__ in, __bf16* __restrict__ vt) {
  __shared__ float tile[32][33];
  int b  = blockIdx.z;
  int s0 = blockIdx.x * 32;
  int d0 = blockIdx.y * 32;
  int tx = threadIdx.x & 31, ty = threadIdx.x >> 5;  // 32 x 8
#pragma unroll
  for (int r = 0; r < 32; r += 8)
    tile[ty + r][tx] = in[((size_t)b * SS + s0 + ty + r) * DIN + d0 + tx];
  __syncthreads();
#pragma unroll
  for (int r = 0; r < 32; r += 8)
    vt[((size_t)b * DIN + d0 + ty + r) * SS + s0 + tx] =
        (__bf16)tile[tx][ty + r];
}

// ---------------------------------------------------------------------------
// NT GEMM: out[m][n] = sum_k A[m][k] * W[n][k] + bias[n], bf16 in/out.
// One wave computes a 16x64 tile (4 WMMA accumulators, K-step 32).
// ---------------------------------------------------------------------------
__global__ __launch_bounds__(128) void gemm_nt_bias(
    const __bf16* __restrict__ A, const __bf16* __restrict__ W,
    const float* __restrict__ bias, __bf16* __restrict__ out,
    int M, int N, int K) {
  int gw   = (blockIdx.x * blockDim.x + threadIdx.x) >> 5;
  int lane = threadIdx.x & 31;
  int ntiles = N >> 6;
  int mt  = gw / ntiles;
  int nt0 = (gw % ntiles) << 6;
  if (mt * 16 >= M) return;
  int half = lane >> 4, ln = lane & 15;

  const __bf16* Arow = A + (size_t)(mt * 16 + ln) * K;
  const __bf16* Wr0  = W + (size_t)(nt0 + ln) * K;

  v8f acc[4] = {};
  for (int ks = 0; ks < K; ks += 32) {
    v16bf av = load_afrag(Arow + ks, half);
#pragma unroll
    for (int t = 0; t < 4; ++t) {
      v16bf bv = load_bfrag(Wr0 + (size_t)t * 16 * K + ks, half);
      acc[t] = __builtin_amdgcn_wmma_f32_16x16x32_bf16(
          false, av, false, bv, (short)0, acc[t], false, false);
    }
  }
#pragma unroll
  for (int t = 0; t < 4; ++t) {
    int n = nt0 + t * 16 + ln;
    float bv = bias[n];
#pragma unroll
    for (int v = 0; v < 8; ++v) {
      int m = mt * 16 + v + (half << 3);   // C/D layout: VGPR v -> row v+8*half
      out[(size_t)m * N + n] = (__bf16)(acc[t][v] + bv);
    }
  }
}

// ---------------------------------------------------------------------------
// Fused label attention: one workgroup = (batch b, 16-label tile).
// Phase A: sim[16][512] = q_tile @ K_b^T  (WMMA, per-wave 64 cols) -> LDS
// Phase B: row softmax (shfl_xor reductions), P bf16 -> LDS
// Phase C: out[16][1024] = P @ V_b via Vt (WMMA, per-wave 128 cols) -> HBM
// ---------------------------------------------------------------------------
__global__ __launch_bounds__(256) void label_attn(
    const __bf16* __restrict__ qbf,   // [L][H]
    const __bf16* __restrict__ Kbf,   // [B][S][H]
    const __bf16* __restrict__ Vt,    // [B][DIN][S]
    const unsigned char* __restrict__ masks,  // [B][S] bool
    float* __restrict__ out) {        // [B][L][DIN]
  __shared__ float  simLDS[16 * SS];     // 32 KB
  __shared__ __bf16 Plds[16 * SS];       // 16 KB

  int b  = blockIdx.y;
  int l0 = blockIdx.x << 4;
  int tid  = threadIdx.x;
  int wave = tid >> 5;
  int lane = tid & 31;
  int half = lane >> 4, ln = lane & 15;

  // ---------------- Phase A: sim tile ----------------
  const __bf16* Qrow = qbf + (size_t)(l0 + ln) * HH;
  const __bf16* Kr0  = Kbf + ((size_t)b * SS + (wave << 6) + ln) * HH;

  v8f acc[4] = {};
  for (int ks = 0; ks < HH; ks += 32) {
    v16bf av = load_afrag(Qrow + ks, half);
#pragma unroll
    for (int nt = 0; nt < 4; ++nt) {
      v16bf bv = load_bfrag(Kr0 + (size_t)nt * 16 * HH + ks, half);
      acc[nt] = __builtin_amdgcn_wmma_f32_16x16x32_bf16(
          false, av, false, bv, (short)0, acc[nt], false, false);
    }
  }
  int s0w = wave << 6;
#pragma unroll
  for (int nt = 0; nt < 4; ++nt) {
    int s = s0w + nt * 16 + ln;
    bool mok = masks[(size_t)b * SS + s] != 0;
#pragma unroll
    for (int v = 0; v < 8; ++v) {
      int m = v + (half << 3);
      simLDS[m * SS + s] = mok ? acc[nt][v] : -3.0e38f;
    }
  }
  __syncthreads();

  // ---------------- Phase B: softmax (wave w owns rows 2w, 2w+1) ----------
  for (int r = wave * 2; r < wave * 2 + 2; ++r) {
    float vals[16];
    float mx = -3.0e38f;
#pragma unroll
    for (int i = 0; i < 16; ++i) {
      float x = simLDS[r * SS + i * 32 + lane];
      vals[i] = x;
      mx = fmaxf(mx, x);
    }
#pragma unroll
    for (int o = 16; o > 0; o >>= 1) mx = fmaxf(mx, __shfl_xor(mx, o, 32));
    float sum = 0.0f;
#pragma unroll
    for (int i = 0; i < 16; ++i) {
      float e = __expf(vals[i] - mx);
      vals[i] = e;
      sum += e;
    }
#pragma unroll
    for (int o = 16; o > 0; o >>= 1) sum += __shfl_xor(sum, o, 32);
    float inv = 1.0f / sum;
#pragma unroll
    for (int i = 0; i < 16; ++i)
      Plds[r * SS + i * 32 + lane] = (__bf16)(vals[i] * inv);
  }
  __syncthreads();

  // ---------------- Phase C: out = P @ V (B-operand from Vt, K=S contig) --
  const __bf16* Prow = Plds + ln * SS;                         // LDS, b128 reads
  const __bf16* Vr0  = Vt + ((size_t)b * DIN + (wave << 7) + ln) * SS;

  v8f oc[8] = {};
  for (int ks = 0; ks < SS; ks += 32) {
    v16bf av = load_afrag(Prow + ks, half);                    // 2x ds_load_b128
#pragma unroll
    for (int nt = 0; nt < 8; ++nt) {
      v16bf bv = load_bfrag(Vr0 + (size_t)nt * 16 * SS + ks, half);
      oc[nt] = __builtin_amdgcn_wmma_f32_16x16x32_bf16(
          false, av, false, bv, (short)0, oc[nt], false, false);
    }
  }
  int n0w = wave << 7;
#pragma unroll
  for (int nt = 0; nt < 8; ++nt) {
    int n = n0w + nt * 16 + ln;
#pragma unroll
    for (int v = 0; v < 8; ++v) {
      int m = v + (half << 3);
      out[((size_t)b * LL + l0 + m) * DIN + n] = oc[nt][v];
    }
  }
}

// ---------------------------------------------------------------------------
extern "C" void kernel_launch(void* const* d_in, const int* in_sizes, int n_in,
                              void* d_out, int out_size, void* d_ws,
                              size_t ws_size, hipStream_t stream) {
  const float*         inputs = (const float*)d_in[0];
  const unsigned char* masks  = (const unsigned char*)d_in[1];
  const float*         emb    = (const float*)d_in[2];
  const float*         Wk     = (const float*)d_in[3];
  const float*         bk     = (const float*)d_in[4];
  const float*         Wq     = (const float*)d_in[5];
  const float*         bq     = (const float*)d_in[6];
  float*               out    = (float*)d_out;

  char* ws = (char*)d_ws;
  // bf16 workspace (~54 MB total; fully L2-resident on MI455X's 192 MB L2)
  __bf16* Vbf = (__bf16*)(ws);                         // B*S*DIN  = 16.78 MB
  __bf16* Kbf = (__bf16*)(ws + (size_t)16777216);      // B*S*H    =  8.39 MB
  __bf16* qbf = (__bf16*)(ws + (size_t)25165824);      // L*H      =  4.19 MB
  __bf16* Wkb = (__bf16*)(ws + (size_t)29360128);      // H*DIN    =  1.05 MB
  __bf16* Wqb = (__bf16*)(ws + (size_t)30408704);      // H*DLBL   =  0.79 MB
  __bf16* Eb  = (__bf16*)(ws + (size_t)31195136);      // L*DLBL   =  6.29 MB
  __bf16* Vt  = (__bf16*)(ws + (size_t)37486592);      // B*DIN*S  = 16.78 MB

  // 1) f32 -> bf16 conversions (vectorized x4)
  cvt_f32_bf16_x4<<<1024, 256, 0, stream>>>(inputs, Vbf, BB * SS * DIN / 4);
  cvt_f32_bf16_x4<<<256,  256, 0, stream>>>(Wk,     Wkb, HH * DIN / 4);
  cvt_f32_bf16_x4<<<256,  256, 0, stream>>>(Wq,     Wqb, HH * DLBL / 4);
  cvt_f32_bf16_x4<<<1024, 256, 0, stream>>>(emb,    Eb,  LL * DLBL / 4);

  // 1b) transposed V for phase C (K-contiguous B operand)
  {
    dim3 g(SS / 32, DIN / 32, BB);
    transpose_to_bf16<<<g, 256, 0, stream>>>(inputs, Vt);
  }

  // 2) K = inputs @ Wk^T + bk : M=8192, N=512, K=1024 -> 4096 waves
  gemm_nt_bias<<<1024, 128, 0, stream>>>(Vbf, Wkb, bk, Kbf, BB * SS, HH, DIN);

  // 3) q = emb @ Wq^T + bq    : M=4096, N=512, K=768  -> 2048 waves
  gemm_nt_bias<<<512, 128, 0, stream>>>(Eb, Wqb, bq, qbf, LL, HH, DLBL);

  // 4) fused attention: grid (L/16, B), 256 threads (8 waves)
  dim3 grid(LL / 16, BB);
  label_attn<<<grid, 256, 0, stream>>>(qbf, Kbf, Vt, masks, out);
}
